// CAT_Transformer_55757265436866
// MI455X (gfx1250) — matrix operand order repaired and checked by reading, not compiled, not run on hardware
//
#include <hip/hip_runtime.h>
#include <hip/hip_bf16.h>
#include <math.h>

// ---------------- model constants ----------------
static constexpr int DIM   = 1024;
static constexpr int NH    = 16;
static constexpr int HD    = 64;
static constexpr int CHUNK = 16;
static constexpr int KCH   = 128;          // number of chunks
static constexpr int BSQ   = 18;           // tokens per fine block (CHUNK+2)
static constexpr int LF    = KCH * BSQ;    // 2304 fine rows
static constexpr int LD    = KCH * BSQ + 2;// 2306 decoder rows
static constexpr int MP    = 2368;         // LD padded to multiple of 64
static constexpr int HID   = 4096;
static constexpr int VOCAB = 32000;
static constexpr float ROPE_C = 0.2878231366f; // ln(10000)/32
static constexpr float EPSV = 1e-5f;

typedef __attribute__((ext_vector_type(16))) _Float16 v16h;
typedef __attribute__((ext_vector_type(8)))  float    v8f;
typedef __attribute__((ext_vector_type(4)))  _Float16 h4_t;

// ---------------- WMMA GEMM: C[M,N] = A[M,K] @ B, f32 in/out, f16 compute ----
// 64x64 block tile, 8 waves (4x2), each wave: 16x32 via two 16x16x32 WMMAs.
// LDS double-buffered; staging is branchless (clamped rows) and 128b-vectorized.
// BT=false: B is row-major [K,N]. BT=true: B is [N,K] (computes A @ B^T).
template<bool BT>
__global__ __launch_bounds__(256)
void gemm_k(const float* __restrict__ A, const float* __restrict__ B,
            float* __restrict__ C, int M, int N, int Kd,
            int lda, int ldb, int ldc) {
  __shared__ _Float16 As[2][64][36];   // [m][k], row stride 72B (8B aligned)
  __shared__ _Float16 Bs[2][64][36];   // [n][k] (transposed in LDS)
  const int n0 = blockIdx.x * 64;
  const int m0 = blockIdx.y * 64;
  const int tid  = threadIdx.x;
  const int wave = tid >> 5, lane = tid & 31;
  const int wm = wave >> 1, wn = wave & 1;     // 4x2 waves -> 64x64 tile
  const int half = lane >> 4, mr = lane & 15;

  auto stage = [&](int buf, int k0) {
    // A tile: 64 rows x 32 cols = 512 float4, 2 per thread, branchless
#pragma unroll
    for (int i = 0; i < 2; ++i) {
      int idx  = tid + i * 256;          // float4 index
      int rowi = idx >> 3, c4 = idx & 7; // 8 float4 per row
      int gr   = m0 + rowi;
      int grc  = gr < M ? gr : M - 1;    // clamp: load always executes
      const float4 va = *(const float4*)(A + (size_t)grc * lda + k0 + c4 * 4);
      h4_t h;
      h[0] = (_Float16)(gr < M ? va.x : 0.f);
      h[1] = (_Float16)(gr < M ? va.y : 0.f);
      h[2] = (_Float16)(gr < M ? va.z : 0.f);
      h[3] = (_Float16)(gr < M ? va.w : 0.f);
      *(h4_t*)&As[buf][rowi][c4 * 4] = h;
    }
    // B tile: 32 k x 64 n = 512 float4, 2 per thread
#pragma unroll
    for (int i = 0; i < 2; ++i) {
      int idx = tid + i * 256;
      if (BT) {
        int nn = idx >> 3, c4 = idx & 7; // contiguous over k
        const float4 vb = *(const float4*)(B + (size_t)(n0 + nn) * ldb + k0 + c4 * 4);
        h4_t h;
        h[0] = (_Float16)vb.x; h[1] = (_Float16)vb.y;
        h[2] = (_Float16)vb.z; h[3] = (_Float16)vb.w;
        *(h4_t*)&Bs[buf][nn][c4 * 4] = h;
      } else {
        int kk = idx >> 4, n4 = idx & 15; // contiguous over n, scatter-transpose
        const float4 vb = *(const float4*)(B + (size_t)(k0 + kk) * ldb + n0 + n4 * 4);
        Bs[buf][n4 * 4 + 0][kk] = (_Float16)vb.x;
        Bs[buf][n4 * 4 + 1][kk] = (_Float16)vb.y;
        Bs[buf][n4 * 4 + 2][kk] = (_Float16)vb.z;
        Bs[buf][n4 * 4 + 3][kk] = (_Float16)vb.w;
      }
    }
  };

  v8f acc0 = {}, acc1 = {};
  stage(0, 0);
  __syncthreads();
  const int nk = Kd >> 5;
  for (int kt = 0; kt < nk; ++kt) {
    const int cur = kt & 1;
    if (kt + 1 < nk) {
      stage(cur ^ 1, (kt + 1) << 5);
      // prefetch the tile after next (global_prefetch_b8 path), branchless
      if (kt + 2 < nk) {
        int gr = m0 + (tid >> 2);
        int grc = gr < M ? gr : M - 1;
        __builtin_prefetch(A + (size_t)grc * lda + ((kt + 2) << 5), 0, 1);
      }
    }
    // pack fragments: A and B both use contiguous k = 16*half + s
    v16h av, bv0, bv1;
#pragma unroll
    for (int vv = 0; vv < 8; ++vv) {
      int kb = 16 * half + 2 * vv;
      av[2 * vv]     = As[cur][wm * 16 + mr][kb];
      av[2 * vv + 1] = As[cur][wm * 16 + mr][kb + 1];
    }
#pragma unroll
    for (int j = 0; j < 16; ++j) {
      bv0[j] = Bs[cur][wn * 32 + mr][16 * half + j];
      bv1[j] = Bs[cur][wn * 32 + 16 + mr][16 * half + j];
    }
    acc0 = __builtin_amdgcn_wmma_f32_16x16x32_f16(false, av, false, bv0, (short)0, acc0, false, false);
    acc1 = __builtin_amdgcn_wmma_f32_16x16x32_f16(false, av, false, bv1, (short)0, acc1, false, false);
    __syncthreads();
  }
  // store: lane col = lane&15, VGPR v -> row v + 8*half
  const int nc0 = n0 + wn * 32 + mr;
#pragma unroll
  for (int vv = 0; vv < 8; ++vv) {
    int mrow = m0 + wm * 16 + vv + 8 * half;
    if (mrow < M) {
      C[(size_t)mrow * ldc + nc0]      = acc0[vv];
      C[(size_t)mrow * ldc + nc0 + 16] = acc1[vv];
    }
  }
}

// ---------------- RMSNorm: one wave per row (cols = 1024) -------------------
__global__ __launch_bounds__(256)
void rmsnorm_k(const float* __restrict__ x, const float* __restrict__ w,
               float* __restrict__ y, int rows) {
  int row  = blockIdx.x * 8 + (threadIdx.x >> 5);
  int lane = threadIdx.x & 31;
  if (row >= rows) return;
  const float* xr = x + (size_t)row * DIM;
  float ss = 0.f;
  for (int c = lane; c < DIM; c += 32) { float v = xr[c]; ss += v * v; }
#pragma unroll
  for (int off = 16; off > 0; off >>= 1) ss += __shfl_xor(ss, off, 32);
  float sc = rsqrtf(ss * (1.f / DIM) + EPSV);
  float* yr = y + (size_t)row * DIM;
  for (int c = lane; c < DIM; c += 32) yr[c] = xr[c] * sc * w[c];
}

// ---------------- elementwise helpers ---------------------------------------
__global__ void add_k(float* __restrict__ x, const float* __restrict__ y, int n) {
  int i = blockIdx.x * blockDim.x + threadIdx.x;
  if (i < n) x[i] += y[i];
}
__global__ void silu_mul_k(float* __restrict__ g, const float* __restrict__ u, int n) {
  int i = blockIdx.x * blockDim.x + threadIdx.x;
  if (i < n) { float v = g[i]; g[i] = (v / (1.f + __expf(-v))) * u[i]; }
}

// ---------------- fine-stage input build ------------------------------------
__global__ void build_fine_k(const int* __restrict__ ids, const int* __restrict__ cspp,
                             const float* __restrict__ f_wte, const float* __restrict__ f_pos,
                             const float* __restrict__ f_adapt, float* __restrict__ X) {
  int idx = blockIdx.x * blockDim.x + threadIdx.x;     // LF*DIM
  if (idx >= LF * DIM) return;
  int r = idx >> 10, d = idx & 1023;
  int ch = r / BSQ, j = r % BSQ;
  float v;
  if (j == 0)      v = f_adapt[cspp[0] * DIM + d];
  else if (j == 1) v = f_pos[(size_t)ch * DIM + d];
  else             v = f_wte[(size_t)ids[ch * CHUNK + (j - 2)] * DIM + d];
  X[idx] = v;
}

// gather rmsnormed rows 2..17 of each chunk, flattened: YC[128][16384]
__global__ void gather_k(const float* __restrict__ H, float* __restrict__ YC) {
  int idx = blockIdx.x * blockDim.x + threadIdx.x;     // KCH*CHUNK*DIM
  if (idx >= KCH * CHUNK * DIM) return;
  int k = idx / (CHUNK * DIM);
  int rem = idx - k * (CHUNK * DIM);
  int c = rem >> 10, d = rem & 1023;
  YC[idx] = H[(size_t)(k * BSQ + 2 + c) * DIM + d];
}

// ---------------- decoder input build (rows padded to MP, pad zeroed) -------
__global__ void build_dec_k(const int* __restrict__ ids, const int* __restrict__ cspp,
                            const float* __restrict__ dummy_fx, const float* __restrict__ FX,
                            const float* __restrict__ sep, const float* __restrict__ wte,
                            float* __restrict__ X) {
  int idx = blockIdx.x * blockDim.x + threadIdx.x;     // MP*DIM
  if (idx >= MP * DIM) return;
  int r = idx >> 10, d = idx & 1023;
  float v = 0.f;
  if (r < LF) {
    int k = r / BSQ, j = r % BSQ;
    if (j == 0)      v = (k == 0) ? dummy_fx[cspp[0] * DIM + d] : FX[(size_t)(k - 1) * DIM + d];
    else if (j == 1) v = sep[d];
    else             v = wte[(size_t)ids[k * CHUNK + (j - 2)] * DIM + d];
  } else if (r == LF)     v = FX[(size_t)(KCH - 1) * DIM + d];
  else if (r == LF + 1)   v = sep[d];
  X[idx] = v;
}

// ---------------- fine attention: one wave per (chunk, head), 18x18 ---------
__global__ __launch_bounds__(32)
void fine_attn_k(const float* __restrict__ QKV, float* __restrict__ O) {
  int ch = blockIdx.x >> 4, h = blockIdx.x & 15;
  __shared__ float Ks[BSQ][HD], Vs[BSQ][HD], Qs[HD], Ss[BSQ], red;
  int lane = threadIdx.x;
  int base = ch * BSQ;
  for (int idx = lane; idx < BSQ * HD; idx += 32) {
    int j = idx >> 6, d = idx & 63;
    const float* kr = QKV + (size_t)(base + j) * (3 * DIM) + DIM + h * HD;
    float x  = kr[d];
    float xp = (d < 32) ? -kr[d + 32] : kr[d - 32];
    float th = __expf(-(float)(d & 31) * ROPE_C);
    float a  = (float)j * th;
    Ks[j][d] = x * __cosf(a) + xp * __sinf(a);
    Vs[j][d] = QKV[(size_t)(base + j) * (3 * DIM) + 2 * DIM + h * HD + d];
  }
  __syncthreads();
  for (int r = 0; r < BSQ; ++r) {
    const float* qr = QKV + (size_t)(base + r) * (3 * DIM) + h * HD;
    for (int d = lane; d < HD; d += 32) {
      float x  = qr[d];
      float xp = (d < 32) ? -qr[d + 32] : qr[d - 32];
      float th = __expf(-(float)(d & 31) * ROPE_C);
      float a  = (float)r * th;
      Qs[d] = x * __cosf(a) + xp * __sinf(a);
    }
    __syncthreads();
    if (lane < BSQ) {
      float s = 0.f;
      for (int d = 0; d < HD; ++d) s += Qs[d] * Ks[lane][d];
      Ss[lane] = s * 0.125f;
    }
    __syncthreads();
    if (lane == 0) {
      float mx = Ss[0];
      for (int j = 1; j < BSQ; ++j) mx = fmaxf(mx, Ss[j]);
      float sum = 0.f;
      for (int j = 0; j < BSQ; ++j) { float e = __expf(Ss[j] - mx); Ss[j] = e; sum += e; }
      red = 1.f / sum;
    }
    __syncthreads();
    float inv = red;
    for (int d = lane; d < HD; d += 32) {
      float a = 0.f;
      for (int j = 0; j < BSQ; ++j) a += Ss[j] * Vs[j][d];
      O[(size_t)(base + r) * DIM + h * HD + d] = a * inv;
    }
    __syncthreads();
  }
}

// ---------------- decoder attention: cat-mask sparse, one wave per (t,h) ----
__global__ __launch_bounds__(32)
void dec_attn_k(const float* __restrict__ QKV, float* __restrict__ O) {
  int t = blockIdx.x >> 4, h = blockIdx.x & 15;
  __shared__ float Qs[HD], Ss[160], red, thl[32];
  __shared__ short KVl[160];
  __shared__ int cnt;
  int lane = threadIdx.x;
  thl[lane] = __expf(-(float)lane * ROPE_C);
  if (lane == 0) {
    int c = 0;
    for (int m = 0; BSQ * m <= t; ++m) KVl[c++] = (short)(BSQ * m);
    int bs0 = (t / BSQ) * BSQ;
    for (int kv = bs0 + 1; kv <= t; ++kv) KVl[c++] = (short)kv;
    cnt = c;
  }
  {
    const float* qr = QKV + (size_t)t * (3 * DIM) + h * HD;
    int pos = t % BSQ;
    for (int d = lane; d < HD; d += 32) {
      float x  = qr[d];
      float xp = (d < 32) ? -qr[d + 32] : qr[d - 32];
      float a  = (float)pos * __expf(-(float)(d & 31) * ROPE_C);
      Qs[d] = x * __cosf(a) + xp * __sinf(a);
    }
  }
  __syncthreads();
  int n = cnt;
  for (int j = lane; j < n; j += 32) {
    int kv = KVl[j];
    int kp = kv % BSQ;
    const float* kr = QKV + (size_t)kv * (3 * DIM) + DIM + h * HD;
    float s = 0.f;
    for (int d = 0; d < HD; ++d) {
      float x  = kr[d];
      float xp = (d < 32) ? -kr[d + 32] : kr[d - 32];
      float a  = (float)kp * thl[d & 31];
      s += Qs[d] * (x * __cosf(a) + xp * __sinf(a));
    }
    Ss[j] = s * 0.125f;
  }
  __syncthreads();
  if (lane == 0) {
    float mx = Ss[0];
    for (int j = 1; j < n; ++j) mx = fmaxf(mx, Ss[j]);
    float sum = 0.f;
    for (int j = 0; j < n; ++j) { float e = __expf(Ss[j] - mx); Ss[j] = e; sum += e; }
    red = 1.f / sum;
  }
  __syncthreads();
  float inv = red;
  for (int d = lane; d < HD; d += 32) {
    float a = 0.f;
    for (int j = 0; j < n; ++j)
      a += Ss[j] * QKV[(size_t)KVl[j] * (3 * DIM) + 2 * DIM + h * HD + d];
    O[(size_t)t * DIM + h * HD + d] = a * inv;
  }
}

// ---------------- host-side helpers -----------------------------------------
static void gemm(hipStream_t s, const float* A, const float* B, float* C,
                 int M, int N, int K, int lda, int ldb, int ldc, bool bt) {
  dim3 g(N / 64, (M + 63) / 64);
  if (bt) gemm_k<true ><<<g, 256, 0, s>>>(A, B, C, M, N, K, lda, ldb, ldc);
  else    gemm_k<false><<<g, 256, 0, s>>>(A, B, C, M, N, K, lda, ldb, ldc);
}
static void rmsnorm(hipStream_t s, const float* x, const float* w, float* y, int rows) {
  rmsnorm_k<<<(rows + 7) / 8, 256, 0, s>>>(x, w, y, rows);
}
static void vadd(hipStream_t s, float* x, const float* y, int n) {
  add_k<<<(n + 255) / 256, 256, 0, s>>>(x, y, n);
}

extern "C" void kernel_launch(void* const* d_in, const int* in_sizes, int n_in,
                              void* d_out, int out_size, void* d_ws, size_t ws_size,
                              hipStream_t stream) {
  (void)in_sizes; (void)n_in; (void)out_size; (void)ws_size;
  const int*   ids      = (const int*)d_in[0];
  const int*   cspp     = (const int*)d_in[1];
  const float* f_wte    = (const float*)d_in[2];
  const float* f_pos    = (const float*)d_in[3];
  const float* f_adapt  = (const float*)d_in[4];
  const float* f_ln1    = (const float*)d_in[5];
  const float* f_ln2    = (const float*)d_in[6];
  const float* f_wqkv   = (const float*)d_in[7];
  const float* f_wo     = (const float*)d_in[8];
  const float* f_wg     = (const float*)d_in[9];
  const float* f_wu     = (const float*)d_in[10];
  const float* f_wd     = (const float*)d_in[11];
  const float* f_norm   = (const float*)d_in[12];
  const float* f_proj   = (const float*)d_in[13];
  const float* dummy_fx = (const float*)d_in[14];
  const float* wte      = (const float*)d_in[15];
  const float* sep      = (const float*)d_in[16];
  const float* dl1      = (const float*)d_in[17];
  const float* dl2      = (const float*)d_in[18];
  const float* d_wqkv   = (const float*)d_in[19];
  const float* d_wo     = (const float*)d_in[20];
  const float* d_wg     = (const float*)d_in[21];
  const float* d_wu     = (const float*)d_in[22];
  const float* d_wd     = (const float*)d_in[23];
  const float* d_norm   = (const float*)d_in[24];
  const float* w_out    = (const float*)d_in[25];
  float* out = (float*)d_out;

  // workspace layout (floats)
  float* ws  = (float*)d_ws;
  float* X   = ws;                         // MP*DIM
  float* H   = X   + (size_t)MP * DIM;     // MP*DIM
  float* QKV = H   + (size_t)MP * DIM;     // MP*3*DIM
  float* O   = QKV + (size_t)MP * 3 * DIM; // MP*DIM
  float* T   = O   + (size_t)MP * DIM;     // MP*DIM
  float* G   = T   + (size_t)MP * DIM;     // MP*HID
  float* U   = G   + (size_t)MP * HID;     // MP*HID
  float* YC  = G;                          // reuse (128*16384 fits)
  float* FX  = U;                          // reuse (128*1024 fits)

  // ===== fine / compressor stage (2304 rows batched) =====
  build_fine_k<<<(LF * DIM + 255) / 256, 256, 0, stream>>>(ids, cspp, f_wte, f_pos, f_adapt, X);
  for (int i = 0; i < 2; ++i) {
    rmsnorm(stream, X, f_ln1 + (size_t)i * DIM, H, LF);
    gemm(stream, H, f_wqkv + (size_t)i * DIM * 3 * DIM, QKV, LF, 3 * DIM, DIM, DIM, 3 * DIM, 3 * DIM, false);
    fine_attn_k<<<KCH * NH, 32, 0, stream>>>(QKV, O);
    gemm(stream, O, f_wo + (size_t)i * DIM * DIM, T, LF, DIM, DIM, DIM, DIM, DIM, false);
    vadd(stream, X, T, LF * DIM);
    rmsnorm(stream, X, f_ln2 + (size_t)i * DIM, H, LF);
    gemm(stream, H, f_wg + (size_t)i * DIM * HID, G, LF, HID, DIM, DIM, HID, HID, false);
    gemm(stream, H, f_wu + (size_t)i * DIM * HID, U, LF, HID, DIM, DIM, HID, HID, false);
    silu_mul_k<<<(LF * HID + 255) / 256, 256, 0, stream>>>(G, U, LF * HID);
    gemm(stream, G, f_wd + (size_t)i * HID * DIM, T, LF, DIM, HID, HID, DIM, DIM, false);
    vadd(stream, X, T, LF * DIM);
  }
  rmsnorm(stream, X, f_norm, H, LF);
  gather_k<<<(KCH * CHUNK * DIM + 255) / 256, 256, 0, stream>>>(H, YC);
  // FX[128,1024] = YC[128,16384] @ f_proj^T  (f_proj is [1024,16384])
  gemm(stream, YC, f_proj, FX, KCH, DIM, CHUNK * DIM, CHUNK * DIM, CHUNK * DIM, DIM, true);

  // ===== decoder stage (2306 rows, padded to 2368) =====
  build_dec_k<<<(MP * DIM + 255) / 256, 256, 0, stream>>>(ids, cspp, dummy_fx, FX, sep, wte, X);
  for (int i = 0; i < 2; ++i) {
    rmsnorm(stream, X, dl1 + (size_t)i * DIM, H, LD);
    gemm(stream, H, d_wqkv + (size_t)i * DIM * 3 * DIM, QKV, LD, 3 * DIM, DIM, DIM, 3 * DIM, 3 * DIM, false);
    dec_attn_k<<<LD * NH, 32, 0, stream>>>(QKV, O);
    gemm(stream, O, d_wo + (size_t)i * DIM * DIM, T, LD, DIM, DIM, DIM, DIM, DIM, false);
    vadd(stream, X, T, LD * DIM);
    rmsnorm(stream, X, dl2 + (size_t)i * DIM, H, LD);
    gemm(stream, H, d_wg + (size_t)i * DIM * HID, G, LD, HID, DIM, DIM, HID, HID, false);
    gemm(stream, H, d_wu + (size_t)i * DIM * HID, U, LD, HID, DIM, DIM, HID, HID, false);
    silu_mul_k<<<(LD * HID + 255) / 256, 256, 0, stream>>>(G, U, LD * HID);
    gemm(stream, G, d_wd + (size_t)i * HID * DIM, T, LD, DIM, HID, HID, DIM, DIM, false);
    vadd(stream, X, T, LD * DIM);
  }
  rmsnorm(stream, X, d_norm, H, LD);
  // logits[2306,32000] = H @ w_out^T  (w_out is [32000,1024])
  gemm(stream, H, w_out, out, LD, VOCAB, DIM, DIM, DIM, VOCAB, true);
}
